// _STN_61916248539270
// MI455X (gfx1250) — matrix operand order repaired and checked by reading, not compiled, not run on hardware
//
#include <hip/hip_runtime.h>
#include <hip/hip_bf16.h>

typedef __attribute__((ext_vector_type(16))) _Float16 v16h;
typedef __attribute__((ext_vector_type(2)))  _Float16 v2h;
typedef __attribute__((ext_vector_type(8)))  float    v8f;
typedef __attribute__((ext_vector_type(2)))  float    v2f;
typedef int v4si __attribute__((vector_size(16)));   // matches builtin param type

#define Bn 32
#define Hn 64
#define Wn 256
#define HW 16384           // Hn*Wn
#define K1FLAT 524288      // Hn*Wn*32 (dense1 K)

#if defined(__gfx1250__) && __has_builtin(__builtin_amdgcn_global_load_async_to_lds_b128) && __has_builtin(__builtin_amdgcn_s_wait_asynccnt)
#define STN_ASYNC_LDS 1
#else
#define STN_ASYNC_LDS 0
#endif

// ---------------- prep kernels ----------------

__global__ void stn_cvt_f16(const float* __restrict__ in, _Float16* __restrict__ outh, int n) {
  int i = blockIdx.x * 256 + threadIdx.x;
  if (i < n) outh[i] = (_Float16)in[i];
}

// pack conv1 weights [5,5,1,16] into WMMA B layout: B elem i of lane l = W[K=16g+i][n=l&15]
__global__ void stn_pack_w1(const float* __restrict__ w1, _Float16* __restrict__ packed) {
  int t = threadIdx.x;            // 512 threads
  int l = t >> 4, i = t & 15;
  int g = l >> 4, n = l & 15;
  int k = 16 * g + i;             // K in 0..31, valid K < 25
  float v = (k < 25) ? w1[k * 16 + n] : 0.f;
  packed[l * 16 + i] = (_Float16)v;
}

// pack conv2 weights [5,5,16,32] into [ntile][13 ksteps][lane][16] B layout
__global__ void stn_pack_w2(const float* __restrict__ w2, _Float16* __restrict__ packed) {
  int t = blockIdx.x * 256 + threadIdx.x;   // 2*13*32*16 = 13312
  if (t >= 13312) return;
  int i = t & 15;
  int l = (t >> 4) & 31;
  int rest = t >> 9;
  int ks = rest % 13;
  int ntile = rest / 13;
  int g = l >> 4, n = l & 15;
  int k = 32 * ks + 16 * g + i;             // K in 0..415, valid < 400
  float v = (k < 400) ? w2[k * 32 + ntile * 16 + n] : 0.f;
  packed[t] = (_Float16)v;
}

__global__ void stn_init_acc(const float* __restrict__ d1b, float* __restrict__ d1acc) {
  int i = blockIdx.x * 256 + threadIdx.x;
  if (i < 2048) d1acc[i] = d1b[i & 63];     // acc[batch*64+n] starts at bias
}

// ---------------- conv1: 5x5, 1->16, one 16x16x32 f16 WMMA per 16-pixel tile ----------------

__global__ __launch_bounds__(128) void stn_conv1_wmma(
    const _Float16* __restrict__ inh, const v16h* __restrict__ packedW1,
    const float* __restrict__ b1, _Float16* __restrict__ a1)
{
  __shared__ _Float16 tile[5 * 68];                  // halo strip: 5 rows x (64+4) cols
  int blk = blockIdx.x;                              // 4 xblks * 64 y * 32 b
  int xblk = blk & 3;
  int y = (blk >> 2) & 63;
  int b = blk >> 8;
  int x0 = xblk * 64;
  int tid = threadIdx.x;

  for (int idx = tid; idx < 5 * 68; idx += 128) {
    int row = idx / 68, col = idx % 68;
    int gy = y + row - 2, gx = x0 + col - 2;
    _Float16 v = (_Float16)0.f;
    if (gy >= 0 && gy < Hn && gx >= 0 && gx < Wn) v = inh[(b * Hn + gy) * Wn + gx];
    tile[idx] = v;
  }
  __syncthreads();

  int lane = tid & 31, wave = tid >> 5;              // wave = mtile (16 pixels each)
  int g = lane >> 4, m = lane & 15, n = lane & 15;

  union { v16h v; _Float16 e[16]; } A;
  for (int i = 0; i < 16; ++i) {
    int k = 16 * g + i;                              // contiguous K layout (matches packed B)
    _Float16 v = (_Float16)0.f;
    if (k < 25) {
      int ky = k / 5, kx = k % 5;
      v = tile[ky * 68 + wave * 16 + m + kx];
    }
    A.e[i] = v;
  }
  v16h Bv = packedW1[lane];
  v8f c = {};
  c = __builtin_amdgcn_wmma_f32_16x16x32_f16(false, A.v, false, Bv, (short)0, c, false, false);

  union { v8f v; float e[8]; } C; C.v = c;
  float bias = b1[n];
  int pixbase = (b * Hn + y) * Wn + x0 + wave * 16;
  for (int r = 0; r < 8; ++r) {
    float val = C.e[r] + bias;
    val = val > 0.f ? val : 0.f;
    a1[(pixbase + r + 8 * g) * 16 + n] = (_Float16)val;
  }
}

// ---------------- conv2: 5x5, 16->32, K=400 -> 13 f16 WMMA k-steps ----------------

__global__ __launch_bounds__(256) void stn_conv2_wmma(
    const _Float16* __restrict__ a1, const v16h* __restrict__ packedW2,
    const float* __restrict__ b2, _Float16* __restrict__ a2)
{
  __shared__ _Float16 tile[5 * 68 * 16];             // 10.9 KB halo tile, 16 channels
  int blk = blockIdx.x;
  int xblk = blk & 3;
  int y = (blk >> 2) & 63;
  int b = blk >> 8;
  int x0 = xblk * 64;
  int tid = threadIdx.x;

  unsigned* tile32 = (unsigned*)tile;
  const unsigned* a132 = (const unsigned*)a1;        // 8 dwords per pixel

#if STN_ASYNC_LDS
  // zero tile (covers SAME-padding halo), then async-DMA the in-bounds 16B chunks
  for (int idx = tid; idx < 5 * 68 * 8; idx += 256) tile32[idx] = 0u;
  __syncthreads();
  for (int idx = tid; idx < 5 * 68 * 2; idx += 256) {       // 2 x b128 per pixel
    int c4 = idx & 1;
    int col = (idx >> 1) % 68;
    int row = idx / (68 * 2);
    int gy = y + row - 2, gx = x0 + col - 2;
    if (gy >= 0 && gy < Hn && gx >= 0 && gx < Wn) {
      v4si* src = (v4si*)(a132 + (((b * Hn + gy) * Wn + gx) * 8 + c4 * 4));
      v4si* dst = (v4si*)(tile32 + ((row * 68 + col) * 8 + c4 * 4));
      __builtin_amdgcn_global_load_async_to_lds_b128(src, dst, 0, 0);
    }
  }
  __builtin_amdgcn_s_wait_asynccnt(0);
  __syncthreads();
#else
  for (int idx = tid; idx < 5 * 68 * 8; idx += 256) {
    int c2 = idx & 7;
    int col = (idx >> 3) % 68;
    int row = idx / (68 * 8);
    int gy = y + row - 2, gx = x0 + col - 2;
    unsigned v = 0u;
    if (gy >= 0 && gy < Hn && gx >= 0 && gx < Wn) v = a132[((b * Hn + gy) * Wn + gx) * 8 + c2];
    tile32[idx] = v;
  }
  __syncthreads();
#endif

  int lane = tid & 31, wave = tid >> 5;
  int mtile = wave & 3, ntile = wave >> 2;           // 4 M-tiles x 2 N-tiles
  int g = lane >> 4, m = lane & 15, n = lane & 15;

  v8f acc = {};
  for (int ks = 0; ks < 13; ++ks) {
    union { v16h v; _Float16 e[16]; } A;
    for (int i = 0; i < 16; ++i) {
      int k = 32 * ks + 16 * g + i;
      _Float16 v = (_Float16)0.f;
      if (k < 400) {
        int ky = k / 80;
        int r2 = k - ky * 80;
        int kx = r2 >> 4, c = r2 & 15;
        v = tile[(ky * 68 + mtile * 16 + m + kx) * 16 + c];
      }
      A.e[i] = v;
    }
    v16h Bv = packedW2[(ntile * 13 + ks) * 32 + lane];
    acc = __builtin_amdgcn_wmma_f32_16x16x32_f16(false, A.v, false, Bv, (short)0, acc, false, false);
  }

  union { v8f v; float e[8]; } C; C.v = acc;
  float bias = b2[ntile * 16 + n];
  int pixbase = (b * Hn + y) * Wn + x0 + mtile * 16;
  for (int r = 0; r < 8; ++r) {
    float val = C.e[r] + bias;
    val = val > 0.f ? val : 0.f;
    a2[(pixbase + r + 8 * g) * 32 + ntile * 16 + n] = (_Float16)val;
  }
}

// ---------------- dense1: M=32, N=64, K=524288 via f32 16x16x4 WMMA, split-K atomics ----------------

__global__ __launch_bounds__(256) void stn_dense1_wmma(
    const _Float16* __restrict__ a2, const float* __restrict__ d1w,
    float* __restrict__ d1acc)
{
  int ntile = blockIdx.x & 3;                        // 4 x 16-col tiles = N=64
  int kc = blockIdx.x >> 2;                          // 64 K-chunks of 8192
  int wave = threadIdx.x >> 5;
  int lane = threadIdx.x & 31;
  int k0 = kc * 8192 + wave * 1024;                  // 1024 K per wave
  int g = lane >> 4, m = lane & 15, n = lane & 15;

  const _Float16* pa0 = a2 + (size_t)m * K1FLAT;         // batches 0..15
  const _Float16* pa1 = a2 + (size_t)(m + 16) * K1FLAT;  // batches 16..31
  const float* pw = d1w + ntile * 16 + n;

  v8f acc0 = {}, acc1 = {};
  for (int k = k0; k < k0 + 1024; k += 4) {
    int kk = k + 2 * g;                              // f32 A: lane-half selects K pair
    v2h h0 = *(const v2h*)(pa0 + kk);
    v2h h1 = *(const v2h*)(pa1 + kk);
    v2f A0; A0.x = (float)h0.x; A0.y = (float)h0.y;
    v2f A1; A1.x = (float)h1.x; A1.y = (float)h1.y;
    v2f Bv;
    Bv.x = pw[(size_t)kk * 64];                      // B VGPR0: K = k+2g
    Bv.y = pw[(size_t)(kk + 1) * 64];                // B VGPR1: K = k+2g+1
    acc0 = __builtin_amdgcn_wmma_f32_16x16x4_f32(false, A0, false, Bv, (short)0, acc0, false, false);
    acc1 = __builtin_amdgcn_wmma_f32_16x16x4_f32(false, A1, false, Bv, (short)0, acc1, false, false);
  }

  union { v8f v; float e[8]; } C0, C1; C0.v = acc0; C1.v = acc1;
  for (int r = 0; r < 8; ++r) {
    atomicAdd(&d1acc[(r + 8 * g) * 64 + ntile * 16 + n], C0.e[r]);
    atomicAdd(&d1acc[(16 + r + 8 * g) * 64 + ntile * 16 + n], C1.e[r]);
  }
}

// ---------------- head: relu(d1acc) @ d2_w + d2_b -> theta[32,6] ----------------

__global__ void stn_head(const float* __restrict__ d1acc, const float* __restrict__ d2w,
                         const float* __restrict__ d2b, float* __restrict__ theta)
{
  int tid = threadIdx.x;
  if (tid >= 192) return;
  int b = tid / 6, i = tid % 6;
  float s = d2b[i];
  for (int j = 0; j < 64; ++j) {
    float x = d1acc[b * 64 + j];
    x = x > 0.f ? x : 0.f;
    s += x * d2w[j * 6 + i];
  }
  theta[b * 6 + i] = s;
}

// ---------------- bilinear sampling transform ----------------

__global__ __launch_bounds__(256) void stn_transform(
    const float* __restrict__ inp, const float* __restrict__ theta, float* __restrict__ out)
{
  int idx = blockIdx.x * 256 + threadIdx.x;          // 524288 pixels
  int b = idx >> 14;
  int p = idx & (HW - 1);
  int i = p >> 8;                                    // row (y)
  int j = p & 255;                                   // col (x)
  float xc = -1.f + 2.f * (float)j / 255.f;          // linspace(-1,1,W)
  float yc = -1.f + 2.f * (float)i / 63.f;           // linspace(-1,1,H)
  const float* th = theta + b * 6;
  float tx = th[0] * xc + th[1] * yc + th[2];
  float ty = th[3] * xc + th[4] * yc + th[5];
  float x = 0.5f * (tx + 1.f) * 256.f;
  float yv = 0.5f * (ty + 1.f) * 64.f;
  int x0 = (int)floorf(x), y0 = (int)floorf(yv);
  int x1 = x0 + 1, y1 = y0 + 1;
  x0 = min(max(x0, 0), Wn - 1); x1 = min(max(x1, 0), Wn - 1);
  y0 = min(max(y0, 0), Hn - 1); y1 = min(max(y1, 0), Hn - 1);
  const float* base = inp + b * HW;
  float pa = base[y0 * Wn + x0];
  float pb = base[y1 * Wn + x0];
  float pc = base[y0 * Wn + x1];
  float pd = base[y1 * Wn + x1];
  float x0f = (float)x0, x1f = (float)x1, y0f = (float)y0, y1f = (float)y1;
  float wa = (x1f - x) * (y1f - yv);
  float wb = (x1f - x) * (yv - y0f);
  float wc = (x - x0f) * (y1f - yv);
  float wd = (x - x0f) * (yv - y0f);
  out[idx] = wa * pa + wb * pb + wc * pc + wd * pd;
}

// ---------------- launch ----------------

extern "C" void kernel_launch(void* const* d_in, const int* in_sizes, int n_in,
                              void* d_out, int out_size, void* d_ws, size_t ws_size,
                              hipStream_t stream) {
  const float* inp  = (const float*)d_in[0];
  const float* w1   = (const float*)d_in[1];
  const float* b1   = (const float*)d_in[2];
  const float* w2   = (const float*)d_in[3];
  const float* b2   = (const float*)d_in[4];
  const float* d1w  = (const float*)d_in[5];
  const float* d1b  = (const float*)d_in[6];
  const float* d2w  = (const float*)d_in[7];
  const float* d2b  = (const float*)d_in[8];
  float* out = (float*)d_out;

  char* ws = (char*)d_ws;
  _Float16* in_h  = (_Float16*)(ws + 0);            // 1 MB
  _Float16* a1    = (_Float16*)(ws + 1048576);      // 16.8 MB
  _Float16* a2    = (_Float16*)(ws + 17825792);     // 33.6 MB
  _Float16* pw1   = (_Float16*)(ws + 51380224);     // 1 KB
  _Float16* pw2   = (_Float16*)(ws + 51381248);     // 26.6 KB
  float*    d1acc = (float*)   (ws + 51407872);     // 8 KB
  float*    theta = (float*)   (ws + 51416064);     // 768 B

  stn_cvt_f16  <<<2048, 256, 0, stream>>>(inp, in_h, Bn * HW);
  stn_pack_w1  <<<1, 512, 0, stream>>>(w1, pw1);
  stn_pack_w2  <<<52, 256, 0, stream>>>(w2, pw2);
  stn_init_acc <<<8, 256, 0, stream>>>(d1b, d1acc);

  stn_conv1_wmma <<<8192, 128, 0, stream>>>(in_h, (const v16h*)pw1, b1, a1);
  stn_conv2_wmma <<<8192, 256, 0, stream>>>(a1, (const v16h*)pw2, b2, a2);
  stn_dense1_wmma<<<256, 256, 0, stream>>>(a2, d1w, d1acc);
  stn_head       <<<1, 192, 0, stream>>>(d1acc, d2w, d2b, theta);
  stn_transform  <<<2048, 256, 0, stream>>>(inp, theta, out);
}
